// AdaAttN_49993419325774
// MI455X (gfx1250) — matrix operand order repaired and checked by reading, compiled, not run on hardware
//
#include <hip/hip_runtime.h>
#include <hip/hip_bf16.h>

// ---------------------------------------------------------------------------
// AdaAttN fused attention for MI455X (gfx1250, wave32, WMMA f32_16x16x32_f16)
// B=4, C=256, H=W=64 (hw=4096). Flash-style: the 16x4096 logit stripe lives
// entirely in CDNA5's 320KB LDS (262KB f32) -> S never touches HBM.
// Normalization folded into affine params (branch-free staging); GEMM output
// layout selected via template (branch-free epilogue).
// ---------------------------------------------------------------------------

typedef __attribute__((ext_vector_type(16))) _Float16 v16h;
typedef __attribute__((ext_vector_type(8)))  _Float16 v8h;
typedef __attribute__((ext_vector_type(8)))  float    v8f;

union H16 { v16h v; v8h h[2]; };

#define B_   4
#define C_   256
#define HW_  4096
#define EPS_ 1e-5f
#define LST  4104   // logit row stride in floats (4096 + 8 pad)
#define NBC  (B_ * C_)

__device__ __forceinline__ v8f wmma_f16(v16h a, v16h b, v8f c) {
  // D = A(16x32 f16) * B(32x16 f16) + C(16x16 f32)
  return __builtin_amdgcn_wmma_f32_16x16x32_f16(false, a, false, b, (short)0, c,
                                                false, false);
}

// A-tile 16(M) x 32(K), f16, from row-major src with row stride `ld` halves.
// ISA layout: lanes 0-15 hold K {0..7,16..23}, lanes 16-31 hold K {8..15,24..31}.
__device__ __forceinline__ v16h load_a_f16(const _Float16* src, int ld, int lane) {
  const int m    = lane & 15;
  const int koff = (lane < 16) ? 0 : 8;
  const _Float16* p = src + m * ld + koff;
  H16 u;
  u.h[0] = *(const v8h*)(p);        // K = koff .. koff+7
  u.h[1] = *(const v8h*)(p + 16);   // K = 16+koff .. 16+koff+7
  return u.v;
}

// B-tile 32(K) x 16(N), f16, from "transposed" row-major srcT[n*ld + k].
// ISA layout: lanes 0-15 hold K {0..7,16..23}, lanes 16-31 hold K {8..15,24..31}.
__device__ __forceinline__ v16h load_b_f16(const _Float16* srcT, int ld, int lane) {
  const int n  = lane & 15;
  const int kb = (lane < 16) ? 0 : 8;
  const _Float16* p = srcT + n * ld + kb;
  H16 u;
  u.h[0] = *(const v8h*)(p);
  u.h[1] = *(const v8h*)(p + 16);
  return u.v;
}

// ---------------------------------------------------------------------------
// Kernel 0: identity affine params (scale=1, shift=0) for the raw-V GEMM.
// ---------------------------------------------------------------------------
__global__ __launch_bounds__(256) void k_ident(float* __restrict__ scale,
                                               float* __restrict__ shift) {
  int i = blockIdx.x * 256 + threadIdx.x;
  if (i < NBC) { scale[i] = 1.0f; shift[i] = 0.0f; }
}

// ---------------------------------------------------------------------------
// Kernel 1: per-(b,c) affine norm params over hw=4096 (ddof=1, +EPS).
//   scale = rstd, shift = -mean*rstd   so that  norm(x) = x*scale + shift
// params layout: [tensor(2)][scale|shift][NBC]
// ---------------------------------------------------------------------------
__global__ __launch_bounds__(256) void k_stats(const float* __restrict__ content,
                                               const float* __restrict__ style,
                                               float* __restrict__ params) {
  const int row = blockIdx.x;  // b*C + c
  const float* src = (blockIdx.y ? style : content) + (size_t)row * HW_;
  float s = 0.f, q = 0.f;
#pragma unroll 4
  for (int i = threadIdx.x; i < HW_; i += 256) {
    float x = src[i];
    s += x; q += x * x;
  }
  for (int m = 16; m; m >>= 1) { s += __shfl_xor(s, m, 32); q += __shfl_xor(q, m, 32); }
  __shared__ float rs[8], rq[8];
  const int wave = threadIdx.x >> 5;
  if ((threadIdx.x & 31) == 0) { rs[wave] = s; rq[wave] = q; }
  __syncthreads();
  if (threadIdx.x == 0) {
    float S = 0.f, Q = 0.f;
    for (int i = 0; i < 8; ++i) { S += rs[i]; Q += rq[i]; }
    float mean = S * (1.0f / HW_);
    float var  = (Q - S * mean) * (1.0f / (HW_ - 1)) + EPS_;  // unbiased + eps
    float rstd = rsqrtf(var);
    float* base = params + (size_t)blockIdx.y * 2 * NBC;
    base[row]       = rstd;          // scale
    base[NBC + row] = -mean * rstd;  // shift
  }
}

// ---------------------------------------------------------------------------
// Kernel 2: fused affine-norm + 1x1 conv GEMM via WMMA.
//   O[o][p] = sum_c W[o][c] * (X[c][p]*scale[c]+shift[c]) + bias[o]
// Block = 8 waves, computes a 16(o) x 128(p) tile; c staged through LDS
// transposed so B-tiles are two contiguous b128 reads per lane.
// TRANS=1 -> out[b][p][o] (for Q and G^T); TRANS=0 -> out[b][o][p] (for V).
// ---------------------------------------------------------------------------
template <int TRANS>
__global__ __launch_bounds__(256) void k_gemm(const float* __restrict__ X,
                                              const float* __restrict__ scale,
                                              const float* __restrict__ shift,
                                              const float* __restrict__ W,
                                              const float* __restrict__ bias,
                                              _Float16* __restrict__ out) {
  const int b  = blockIdx.z;
  const int p0 = blockIdx.x * 128;
  const int o0 = blockIdx.y * 16;
  const int tid = threadIdx.x, lane = tid & 31, wave = tid >> 5;
  __shared__ _Float16 xs[128 * 40];  // [p][c_local] stride 40 (80B rows, 16B aligned)
  const float* Xb = X + (size_t)b * C_ * HW_;
  v8f acc = {};

  for (int cc = 0; cc < 8; ++cc) {   // K loop: c chunks of 32
    __syncthreads();
    // stage [32 c][128 p], affine-normalized, f16, transposed (coalesced reads)
    {
      const int cl0 = tid >> 7;          // this thread's c rows: cl0, cl0+2, ...
      const int pl  = tid & 127;
#pragma unroll
      for (int i = 0; i < 16; ++i) {
        int cl = cl0 + i * 2;
        int c  = cc * 32 + cl;
        float x = Xb[(size_t)c * HW_ + p0 + pl];
        x = fmaf(x, scale[b * C_ + c], shift[b * C_ + c]);
        xs[pl * 40 + cl] = (_Float16)x;
      }
    }
    __syncthreads();
    // A-tile: weight rows o0..o0+15, f32 -> f16 inline (two 32B groups per lane)
    {
      const int m = lane & 15, koff = (lane < 16) ? 0 : 8;
      const float* wr = W + (size_t)(o0 + m) * C_ + cc * 32 + koff;
      H16 a;
#pragma unroll
      for (int v = 0; v < 8; ++v) {
        a.v[v]     = (_Float16)wr[v];
        a.v[8 + v] = (_Float16)wr[16 + v];
      }
      v16h bt = load_b_f16(xs + wave * 16 * 40, 40, lane);
      acc = wmma_f16(a.v, bt, acc);
    }
  }

  // Branch-free epilogue: compile-time layout, constant strides.
  const int n = lane & 15, mh = (lane >> 4) << 3;
  const int oB = o0 + mh;                 // this lane's first o row (r=0)
  const int p  = p0 + wave * 16 + n;      // this lane's pixel column
  _Float16* ob;
  if (TRANS) ob = out + (size_t)b * HW_ * C_ + (size_t)p * C_ + oB;   // stride 1 in r
  else       ob = out + (size_t)b * C_ * HW_ + (size_t)oB * HW_ + p;  // stride HW_ in r
#pragma unroll
  for (int r = 0; r < 8; ++r) {
    float v = acc[r] + bias[oB + r];
    ob[TRANS ? r : r * HW_] = (_Float16)v;
  }
}

// ---------------------------------------------------------------------------
// Kernel 3: fused attention. One block per (b, 16-query stripe), 8 waves.
//  Phase 1: logits[16][4096] = Q*G into 262KB LDS (f32), tiled across waves.
//  Phase 2: row softmax (16 lanes/row, shfl reductions), exp in place.
//  Phase 3: P*V and P*(V.V); each wave owns 32 output channels; 1/rowsum
//           folded into epilogue; mean/std written straight to d_out.
// ---------------------------------------------------------------------------
__global__ __launch_bounds__(256) void k_attn(const _Float16* __restrict__ Q,   // [b][p][c]
                                              const _Float16* __restrict__ GT,  // [b][k][c]
                                              const _Float16* __restrict__ VT,  // [b][o][k]
                                              float* __restrict__ meanOut,
                                              float* __restrict__ stdOut) {
  extern __shared__ float lds[];
  float* ldsL     = lds;             // [16][LST]
  float* ldsRecip = lds + 16 * LST;  // [16]
  const int b = blockIdx.y, q0 = blockIdx.x * 16;
  const int tid = threadIdx.x, lane = tid & 31, wave = tid >> 5;

  // Q A-tiles (16 queries x full 256 channels) resident in 64 VGPRs
  v16h qa[8];
  const _Float16* Qb = Q + ((size_t)b * HW_ + q0) * C_;
#pragma unroll
  for (int cc = 0; cc < 8; ++cc) qa[cc] = load_a_f16(Qb + cc * 32, C_, lane);

  // ---- Phase 1: logits ----
  const _Float16* Gb = GT + (size_t)b * HW_ * C_;
  const int nn = lane & 15, mh = (lane >> 4) << 3;
  for (int kt = wave; kt < 256; kt += 8) {   // 16-key tiles, wave-strided
    v8f s = {};
    const _Float16* gp0 = Gb + (size_t)(kt * 16 + nn) * C_ + ((lane < 16) ? 0 : 8);
#pragma unroll
    for (int cc = 0; cc < 8; ++cc) {
      H16 bb;
      const _Float16* gp = gp0 + cc * 32;
      bb.h[0] = *(const v8h*)(gp);
      bb.h[1] = *(const v8h*)(gp + 16);
      s = wmma_f16(qa[cc], bb.v, s);
    }
#pragma unroll
    for (int r = 0; r < 8; ++r)
      ldsL[(size_t)(r + mh) * LST + kt * 16 + nn] = s[r];
  }
  __syncthreads();

  // ---- Phase 2: softmax, 16 lanes per row ----
  {
    const int row = tid >> 4, l16 = tid & 15;
    float* L = ldsL + (size_t)row * LST;
    float mx = -3.0e38f;
#pragma unroll 4
    for (int j = l16; j < HW_; j += 16) mx = fmaxf(mx, L[j]);
    for (int m = 8; m; m >>= 1) mx = fmaxf(mx, __shfl_xor(mx, m, 32));
    float sum = 0.f;
#pragma unroll 4
    for (int j = l16; j < HW_; j += 16) {
      float e = __expf(L[j] - mx);
      L[j] = e;
      sum += e;
    }
    for (int m = 8; m; m >>= 1) sum += __shfl_xor(sum, m, 32);
    if (l16 == 0) ldsRecip[row] = 1.0f / sum;
  }
  __syncthreads();

  // ---- Phase 3: P*V and P*(V.V) ----
  const _Float16* Vb = VT + (size_t)b * C_ * HW_;
  v8f am0 = {}, am1 = {}, as0 = {}, as1 = {};
  const int m    = lane & 15;
  const int koff = (lane < 16) ? 0 : 8;
  const int kb   = (lane < 16) ? 0 : 8;
  const _Float16* vrow0 = Vb + (size_t)(wave * 32 + nn) * HW_;       // o tile 0
  const _Float16* vrow1 = Vb + (size_t)(wave * 32 + 16 + nn) * HW_;  // o tile 1
  for (int kc = 0; kc < 128; ++kc) {   // key chunks of 32
    // P A-tile: f32 LDS -> f16 registers (two 8-float contiguous groups)
    H16 pa;
    const float* pr = ldsL + (size_t)m * LST + kc * 32;
#pragma unroll
    for (int v = 0; v < 8; ++v) {
      pa.v[v]     = (_Float16)pr[koff + v];
      pa.v[8 + v] = (_Float16)pr[16 + koff + v];
    }
    H16 b0, b1;
    const _Float16* p0 = vrow0 + kc * 32 + kb;
    const _Float16* p1 = vrow1 + kc * 32 + kb;
    b0.h[0] = *(const v8h*)(p0); b0.h[1] = *(const v8h*)(p0 + 16);
    b1.h[0] = *(const v8h*)(p1); b1.h[1] = *(const v8h*)(p1 + 16);
    am0 = wmma_f16(pa.v, b0.v, am0);
    am1 = wmma_f16(pa.v, b1.v, am1);
    v16h sq0 = b0.v * b0.v;   // packed f16 squares for the second moment
    v16h sq1 = b1.v * b1.v;
    as0 = wmma_f16(pa.v, sq0, as0);
    as1 = wmma_f16(pa.v, sq1, as1);
  }

  // ---- Epilogue: mean/std -> d_out [b][c][h][w] ----
#pragma unroll
  for (int r = 0; r < 8; ++r) {
    const int M = r + mh;
    const float rc = ldsRecip[M];
    const int q = q0 + M;
#pragma unroll
    for (int t = 0; t < 2; ++t) {
      int o = wave * 32 + t * 16 + nn;
      float mn  = (t ? am1[r] : am0[r]) * rc;
      float sec = (t ? as1[r] : as0[r]) * rc;
      float sd  = sqrtf(fmaxf(sec - mn * mn, 0.f));
      size_t idx = ((size_t)b * C_ + o) * HW_ + q;
      meanOut[idx] = mn;
      stdOut[idx]  = sd;
    }
  }
}

// ---------------------------------------------------------------------------
// Kernel 4: out = std * (content*scaleC + shiftC) + mean   (elementwise)
// ---------------------------------------------------------------------------
__global__ __launch_bounds__(256) void k_final(const float* __restrict__ content,
                                               const float* __restrict__ cparams,
                                               const float* __restrict__ meanOut,
                                               const float* __restrict__ stdOut,
                                               float* __restrict__ out) {
  const int i0 = (blockIdx.x * 256 + threadIdx.x) * 4;
  const int bc = i0 >> 12;                    // i0 / HW_
  const float sc = cparams[bc];
  const float sh = cparams[NBC + bc];
#pragma unroll
  for (int j = 0; j < 4; ++j) {
    int i = i0 + j;
    float xn = fmaf(content[i], sc, sh);
    out[i] = fmaf(stdOut[i], xn, meanOut[i]);
  }
}

// ---------------------------------------------------------------------------
extern "C" void kernel_launch(void* const* d_in, const int* in_sizes, int n_in,
                              void* d_out, int out_size, void* d_ws, size_t ws_size,
                              hipStream_t stream) {
  const float* content = (const float*)d_in[0];
  const float* style   = (const float*)d_in[1];
  const float* Wf = (const float*)d_in[2];
  const float* bf = (const float*)d_in[3];
  const float* Wg = (const float*)d_in[4];
  const float* bg = (const float*)d_in[5];
  const float* Wh = (const float*)d_in[6];
  const float* bh = (const float*)d_in[7];

  float* out = (float*)d_out;
  const int OUTN = B_ * C_ * HW_;        // 4,194,304
  float* meanOut = out + OUTN;           // tuple order: (out, mean, std)
  float* stdOut  = out + 2 * OUTN;

  // workspace: params (64KB region) + Q/GT/VT f16 (8MB each)
  char* ws = (char*)d_ws;
  float*    params = (float*)ws;                       // [scaleC,shiftC,scaleS,shiftS,scaleI,shiftI][NBC]
  _Float16* wsQ   = (_Float16*)(ws + 65536);           // [b][p][c]
  _Float16* wsGT  = wsQ  + (size_t)B_ * HW_ * C_;      // [b][k][c]
  _Float16* wsVT  = wsGT + (size_t)B_ * HW_ * C_;      // [b][o][k]

  float* scaleC = params;            // content: scale, shift
  float* shiftC = params + NBC;
  float* scaleS = params + 2 * NBC;  // style: scale, shift
  float* shiftS = params + 3 * NBC;
  float* scaleI = params + 4 * NBC;  // identity (for raw V)
  float* shiftI = params + 5 * NBC;

  k_ident<<<dim3((NBC + 255) / 256), 256, 0, stream>>>(scaleI, shiftI);
  k_stats<<<dim3(NBC, 2), 256, 0, stream>>>(content, style, params);

  dim3 ggrid(HW_ / 128, C_ / 16, B_);
  k_gemm<1><<<ggrid, 256, 0, stream>>>(content, scaleC, shiftC, Wf, bf, wsQ);
  k_gemm<1><<<ggrid, 256, 0, stream>>>(style,   scaleS, shiftS, Wg, bg, wsGT);
  k_gemm<0><<<ggrid, 256, 0, stream>>>(style,   scaleI, shiftI, Wh, bh, wsVT);

  const size_t ldsBytes = (size_t)(16 * LST + 16) * sizeof(float);  // ~262.7KB
  k_attn<<<dim3(HW_ / 16, B_), 256, ldsBytes, stream>>>(wsQ, wsGT, wsVT,
                                                        meanOut, stdOut);

  k_final<<<dim3(OUTN / 1024), 256, 0, stream>>>(content, params, meanOut, stdOut, out);
}
